// LSTMDecoder_28845000360517
// MI455X (gfx1250) — matrix operand order repaired and checked by reading, not compiled, not run on hardware
//
#include <hip/hip_runtime.h>
#include <hip/hip_bf16.h>

// ---------------------------------------------------------------------------
// LSTM attention decoder for MI455X (gfx1250, wave32, WMMA).
// All GEMMs use v_wmma_f32_16x16x32_bf16 (bf16 A/B, fp32 accumulate).
// Vocab GEMM (134 GFLOP, dominant) stages macro-tiles into LDS with the
// CDNA5 async copy engine (global_load_async_to_lds_b128 + s_wait_asynccnt),
// consumes fragments via ds_load_b128, and prefetches the next K-chunk.
// ---------------------------------------------------------------------------

typedef __attribute__((ext_vector_type(16))) __bf16 v16bf;
typedef __attribute__((ext_vector_type(8)))  float  v8f;

struct alignas(16) Q16 { unsigned long long x, y; };   // 16-byte POD chunk
union AFrag { v16bf v; Q16 q[2]; };

#define BB   32      // batch
#define SS   128     // source length
#define TT   128     // target steps (T_TGT - 1)
#define EE   256     // embedding dim
#define HH   512     // hidden
#define VV   32000   // vocab
#define KG   1280    // E + H + H  (fused gates K)
#define NG   2048    // 4H
#define KU   1536    // 3H (context 2H + h)
#define K2H  1024    // 2H

// ------------------------- workspace layout (bytes) ------------------------
#define OFF_WG    0UL                         // 2048*1280 bf16  = 5242880
#define OFF_BG    5242880UL                   // 2048 f32        = 8192
#define OFF_WATT  5251072UL                   // 512*1024 bf16   = 1048576
#define OFF_ENCB  6299648UL                   // 32*128*1024 bf16= 8388608
#define OFF_WDEC  14688256UL                  // 512*1536 bf16   = 1572864
#define OFF_WVOC  16261120UL                  // 32000*512 bf16  = 32768000
#define OFF_KEYS  49029120UL                  // 32*128*512 f32  = 8388608
#define OFF_H     57417728UL                  // 32*512 f32      = 65536
#define OFF_C     57483264UL                  // 32*512 f32      = 65536
#define OFF_OBF   57548800UL                  // 32*512 bf16     = 32768
#define OFF_XBAR  57581568UL                  // 32*1280 bf16    = 81920
#define OFF_G     57663488UL                  // 32*2048 f32     = 262144
#define OFF_U     57925632UL                  // 32*1536 bf16    = 98304
#define OFF_OALL  58023936UL                  // 128*32*512 bf16 = 4194304
// total = 62218240 bytes (~62.3 MB)

// CDNA5 async copy: per-lane 16B  MEM[gaddr] -> LDS[ldsaddr]; ASYNCcnt-tracked.
__device__ __forceinline__ void async_copy_b128(unsigned ldsaddr, const void* gaddr) {
    asm volatile("global_load_async_to_lds_b128 %0, %1, off"
                 :: "v"(ldsaddr), "v"(gaddr) : "memory");
}
__device__ __forceinline__ void async_wait0() {
    asm volatile("s_wait_asynccnt 0x0" ::: "memory");
}
__device__ __forceinline__ unsigned lds_addr32(const void* p) {
    // LDS aperture: flat addr[31:0] is the LDS byte offset (ISA 10.2)
    return (unsigned)(unsigned long long)p;
}

// ---------------------------------------------------------------------------
// Generic bf16 WMMA GEMM:  C[M,N] = act( A[M,K] * B[N,K]^T + bias[N] )
// One 16x16 D-tile per wave; fragments straight from L2 (latency path for
// the tiny recurrent GEMMs). EXEC all ones in every wave.
// ---------------------------------------------------------------------------
__global__ __launch_bounds__(256) void wmma_gemm_bf16(
    const __bf16* __restrict__ A, int lda,
    const __bf16* __restrict__ B, int K,
    float* __restrict__ C, __bf16* __restrict__ Cb0, __bf16* __restrict__ Cb1,
    int ldc, const float* __restrict__ bias,
    int Mtiles, int Ntiles, int act)
{
    int tile = blockIdx.x * 8 + (threadIdx.x >> 5);
    if (tile >= Mtiles * Ntiles) return;            // wave-uniform
    int tm = tile / Ntiles, tn = tile % Ntiles;
    int lane = threadIdx.x & 31;
    int lr = lane & 15, hf = lane >> 4;

    const __bf16* arow = A + (long)(tm * 16 + lr) * lda;   // A row for this lane
    const __bf16* brow = B + (long)(tn * 16 + lr) * K;     // weight row = column n
    v8f acc = {};

    for (int k = 0; k < K; k += 32) {
        AFrag fa, fb;
        // A 16x32 bf16 layout: elems0-7 = K(hf*8 .. +7), elems8-15 = K(16+hf*8 ..)
        fa.q[0] = *(const Q16*)(arow + k + hf * 8);
        fa.q[1] = *(const Q16*)(arow + k + 16 + hf * 8);
        // B 32x16 bf16 layout: elems0-7 = K(hf*8 .. +7), elems8-15 = K(16+hf*8 ..)
        fb.q[0] = *(const Q16*)(brow + k + hf * 8);
        fb.q[1] = *(const Q16*)(brow + k + 16 + hf * 8);
        acc = __builtin_amdgcn_wmma_f32_16x16x32_bf16(
                  false, fa.v, false, fb.v, (short)0, acc, false, false);
    }

    float bv = bias ? bias[tn * 16 + lr] : 0.0f;
#pragma unroll
    for (int r = 0; r < 8; ++r) {
        float v = acc[r] + bv;
        if (act) v = tanhf(v);
        long off = (long)(tm * 16 + hf * 8 + r) * ldc + (tn * 16 + lr);
        if (C)   C[off]   = v;
        if (Cb0) Cb0[off] = (__bf16)v;
        if (Cb1) Cb1[off] = (__bf16)v;
    }
}

// ---------------------------------------------------------------------------
// Vocab GEMM (throughput path): logits[b*T+t, v] = o_all[t*B+b] . Wv[v] + bias
// M=4096, N=32000, K=512.
// Block = 256 threads (8 waves) computes a 32x128 macro-tile:
//   async-copy A chunk (32xKC, shared by 4 waves) and B chunk (128xKC, shared
//   by 2 waves) into LDS; each wave computes 1 M-tile x 2 N-tiles with the A
//   fragment reused. Rows padded 16B to break the 128B-stride bank conflict.
// ---------------------------------------------------------------------------
#define MC 32
#define NC 128
#define KC 64
#define LPAD 8   // pad in bf16 elements (16B) per row

__global__ __launch_bounds__(256) void wmma_gemm_vocab_lds(
    const __bf16* __restrict__ Oall,     // [t][b][H] bf16
    const __bf16* __restrict__ Wv,       // [V][H] bf16
    const float*  __restrict__ bvoc,
    float* __restrict__ out)             // [b][t][V] f32
{
    __shared__ __bf16 lsA[MC][KC + LPAD];   // 4.5 KB
    __shared__ __bf16 lsB[NC][KC + LPAD];   // 18 KB
    const int NB = VV / NC;                 // 250
    int mb = blockIdx.x / NB;               // 0..127
    int nb = blockIdx.x % NB;               // 0..249
    int m0 = mb * MC, n0 = nb * NC;

    int tid  = threadIdx.x;
    int wave = tid >> 5, lane = tid & 31;
    int lr = lane & 15, hf = lane >> 4;
    int tmw = wave >> 2;                    // 0..1   (M-tile of this wave)
    int tnw = (wave & 3) * 2;               // 0,2,4,6 (N-tile pair base)

    // A staging assignment: 32 rows x (KC*2B=128B) = 256 chunks of 16B
    int arow = tid >> 3;                    // 0..31
    int aseg = tid & 7;                     // 0..7
    int am   = m0 + arow;                   // logical row = b*128 + t
    const __bf16* agsrc = Oall + ((long)(am & 127) * BB + (am >> 7)) * HH;
    unsigned alds = lds_addr32(&lsA[arow][aseg * 8]);

    v8f acc0 = {}, acc1 = {};

    for (int k0 = 0; k0 < HH; k0 += KC) {
        // ---- async stage A (32xKC): 1 x b128 per lane ----
        async_copy_b128(alds, agsrc + k0 + aseg * 8);
        // ---- async stage B (128xKC): 4 x b128 per lane ----
#pragma unroll
        for (int r = 0; r < 4; ++r) {
            int idx  = tid + r * 256;
            int brow = idx >> 3;            // 0..127
            int bseg = idx & 7;             // 0..7
            async_copy_b128(lds_addr32(&lsB[brow][bseg * 8]),
                            Wv + (long)(n0 + brow) * HH + k0 + bseg * 8);
        }
        // warm next K-chunk while this one computes (global_prefetch_b8)
        if (k0 + KC < HH) {
            __builtin_prefetch(agsrc + k0 + KC, 0, 3);
            __builtin_prefetch(Wv + (long)(n0 + (tid >> 1)) * HH + k0 + KC, 0, 3);
        }
        async_wait0();            // wave's ASYNCcnt -> 0 (its LDS writes landed)
        __syncthreads();          // cross-wave visibility
        // ---- compute: 2 WMMA k-steps per chunk, 2 N-tiles per wave ----
#pragma unroll
        for (int kk = 0; kk < KC; kk += 32) {
            AFrag fa, fb0, fb1;
            fa.q[0]  = *(const Q16*)(&lsA[tmw * 16 + lr][kk + hf * 8]);
            fa.q[1]  = *(const Q16*)(&lsA[tmw * 16 + lr][kk + 16 + hf * 8]);
            fb0.q[0] = *(const Q16*)(&lsB[tnw * 16 + lr][kk + hf * 8]);
            fb0.q[1] = *(const Q16*)(&lsB[tnw * 16 + lr][kk + 16 + hf * 8]);
            fb1.q[0] = *(const Q16*)(&lsB[(tnw + 1) * 16 + lr][kk + hf * 8]);
            fb1.q[1] = *(const Q16*)(&lsB[(tnw + 1) * 16 + lr][kk + 16 + hf * 8]);
            acc0 = __builtin_amdgcn_wmma_f32_16x16x32_bf16(
                       false, fa.v, false, fb0.v, (short)0, acc0, false, false);
            acc1 = __builtin_amdgcn_wmma_f32_16x16x32_bf16(
                       false, fa.v, false, fb1.v, (short)0, acc1, false, false);
        }
        __syncthreads();          // protect LDS against next chunk's overwrite
    }

    float bv0 = bvoc[n0 + tnw * 16 + lr];
    float bv1 = bvoc[n0 + (tnw + 1) * 16 + lr];
#pragma unroll
    for (int r = 0; r < 8; ++r) {
        long row = m0 + tmw * 16 + hf * 8 + r;        // = b*128 + t, matches out
        out[row * (long)VV + (n0 + tnw * 16 + lr)]       = acc0[r] + bv0;
        out[row * (long)VV + (n0 + (tnw + 1) * 16 + lr)] = acc1[r] + bv1;
    }
}

// ---------------------------- pointwise kernels ----------------------------
__global__ void f2bf(const float* __restrict__ s, __bf16* __restrict__ d, long n) {
    long i = blockIdx.x * (long)blockDim.x + threadIdx.x;
    long st = (long)gridDim.x * blockDim.x;
    for (; i < n; i += st) d[i] = (__bf16)s[i];
}

__global__ void prep_wg(const float* __restrict__ Wih, const float* __restrict__ Whh,
                        __bf16* __restrict__ Wg) {
    long n = (long)NG * KG;
    long st = (long)gridDim.x * blockDim.x;
    for (long i = blockIdx.x * (long)blockDim.x + threadIdx.x; i < n; i += st) {
        long j = i / KG, k = i % KG;
        float v = (k < (EE + HH)) ? Wih[j * (EE + HH) + k] : Whh[j * HH + (k - EE - HH)];
        Wg[i] = (__bf16)v;
    }
}

__global__ void prep_bg(const float* bih, const float* bhh, float* bg) {
    int i = blockIdx.x * blockDim.x + threadIdx.x;
    if (i < NG) bg[i] = bih[i] + bhh[i];
}

__global__ void init_state(const float* h0, const float* c0,
                           float* h, float* c, __bf16* obf) {
    int i = blockIdx.x * blockDim.x + threadIdx.x;
    if (i >= BB * HH) return;
    h[i] = h0[i]; c[i] = c0[i]; obf[i] = (__bf16)0.0f;
}

// x_bar = [ emb[token] | o_prev | h_prev ] as bf16, one row per batch
__global__ __launch_bounds__(256) void pack_xbar(
    const int* __restrict__ ids, const float* __restrict__ emb,
    const __bf16* __restrict__ obf, const float* __restrict__ h,
    __bf16* __restrict__ xb, int t)
{
    int b = blockIdx.x, tid = threadIdx.x;
    int tok = ids[b * (TT + 1) + t];
    __bf16* row = xb + (long)b * KG;
    if (tid < EE) row[tid] = (__bf16)emb[(long)tok * EE + tid];
    for (int j = tid; j < HH; j += 256) row[EE + j]      = obf[b * HH + j];
    for (int j = tid; j < HH; j += 256) row[EE + HH + j] = (__bf16)h[b * HH + j];
}

__global__ void lstm_cell(const float* __restrict__ G, float* h, float* c) {
    int idx = blockIdx.x * blockDim.x + threadIdx.x;
    if (idx >= BB * HH) return;
    int b = idx >> 9, j = idx & 511;
    const float* g = G + (long)b * NG;
    float ig = g[j], fg = g[HH + j], gg = g[2 * HH + j], og = g[3 * HH + j];
    float si = 1.0f / (1.0f + __expf(-ig));
    float sf = 1.0f / (1.0f + __expf(-fg));
    float so = 1.0f / (1.0f + __expf(-og));
    float cn = sf * c[idx] + si * tanhf(gg);
    float hn = so * tanhf(cn);
    c[idx] = cn; h[idx] = hn;
}

// attention: scores -> softmax -> context; build u = [a | h] bf16
__global__ __launch_bounds__(128) void attn_ctx(
    const float* __restrict__ h, const float* __restrict__ keys,
    const float* __restrict__ enc, const int* __restrict__ mask,
    __bf16* __restrict__ u)
{
    int b = blockIdx.x, tid = threadIdx.x;        // 128 threads, 1 per source pos
    __shared__ float at[SS];
    __shared__ float red[SS];
    const float* hp = h + b * HH;
    const float* kp = keys + ((long)b * SS + tid) * HH;
    float sc = 0.0f;
    for (int k = 0; k < HH; ++k) sc += hp[k] * kp[k];
    if (mask[b * SS + tid] <= 0) sc += -1.0e9f;
    at[tid] = sc; red[tid] = sc; __syncthreads();
    for (int o = 64; o; o >>= 1) { if (tid < o) red[tid] = fmaxf(red[tid], red[tid + o]); __syncthreads(); }
    float mx = red[0]; __syncthreads();
    float e = __expf(at[tid] - mx);
    red[tid] = e; __syncthreads();
    for (int o = 64; o; o >>= 1) { if (tid < o) red[tid] += red[tid + o]; __syncthreads(); }
    float inv = 1.0f / red[0]; __syncthreads();
    at[tid] = e * inv; __syncthreads();

    for (int k = tid; k < K2H; k += 128) {
        float a = 0.0f;
        for (int s = 0; s < SS; ++s) a += at[s] * enc[((long)b * SS + s) * K2H + k];
        u[(long)b * KU + k] = (__bf16)a;
    }
    for (int j = tid; j < HH; j += 128) u[(long)b * KU + K2H + j] = (__bf16)hp[j];
}

// in-place log-softmax over 32000-wide rows
__global__ __launch_bounds__(256) void log_softmax_rows(float* __restrict__ x) {
    float* p = x + (long)blockIdx.x * VV;
    __shared__ float red[256];
    int tid = threadIdx.x;
    float mx = -3.4e38f;
    for (int i = tid; i < VV; i += 256) mx = fmaxf(mx, p[i]);
    red[tid] = mx; __syncthreads();
    for (int o = 128; o; o >>= 1) { if (tid < o) red[tid] = fmaxf(red[tid], red[tid + o]); __syncthreads(); }
    mx = red[0]; __syncthreads();
    float s = 0.0f;
    for (int i = tid; i < VV; i += 256) s += __expf(p[i] - mx);
    red[tid] = s; __syncthreads();
    for (int o = 128; o; o >>= 1) { if (tid < o) red[tid] += red[tid + o]; __syncthreads(); }
    float lse = mx + __logf(red[0]);
    for (int i = tid; i < VV; i += 256) p[i] = p[i] - lse;
}

// ---------------------------------------------------------------------------
extern "C" void kernel_launch(void* const* d_in, const int* in_sizes, int n_in,
                              void* d_out, int out_size, void* d_ws, size_t ws_size,
                              hipStream_t stream) {
    const float* enc  = (const float*)d_in[0];
    const int*   mask = (const int*)  d_in[1];
    const int*   ids  = (const int*)  d_in[2];
    const float* h0   = (const float*)d_in[3];
    const float* c0   = (const float*)d_in[4];
    const float* emb  = (const float*)d_in[5];
    const float* Wih  = (const float*)d_in[6];
    const float* bih  = (const float*)d_in[7];
    const float* Whh  = (const float*)d_in[8];
    const float* bhh  = (const float*)d_in[9];
    const float* Watt = (const float*)d_in[10];
    const float* Wdec = (const float*)d_in[11];
    const float* bdec = (const float*)d_in[12];
    const float* Wvoc = (const float*)d_in[13];
    const float* bvoc = (const float*)d_in[14];

    char* ws = (char*)d_ws;
    __bf16* Wg     = (__bf16*)(ws + OFF_WG);
    float*  bg     = (float*) (ws + OFF_BG);
    __bf16* WattB  = (__bf16*)(ws + OFF_WATT);
    __bf16* encB   = (__bf16*)(ws + OFF_ENCB);
    __bf16* WdecB  = (__bf16*)(ws + OFF_WDEC);
    __bf16* WvocB  = (__bf16*)(ws + OFF_WVOC);
    float*  keys   = (float*) (ws + OFF_KEYS);
    float*  h      = (float*) (ws + OFF_H);
    float*  c      = (float*) (ws + OFF_C);
    __bf16* obf    = (__bf16*)(ws + OFF_OBF);
    __bf16* xbar   = (__bf16*)(ws + OFF_XBAR);
    float*  G      = (float*) (ws + OFF_G);
    __bf16* u      = (__bf16*)(ws + OFF_U);
    __bf16* oall   = (__bf16*)(ws + OFF_OALL);
    float*  out    = (float*)d_out;

    // ---- one-time prep (re-run every call; deterministic) ----
    f2bf<<<512,  256, 0, stream>>>(Watt, WattB, (long)HH * K2H);
    f2bf<<<1024, 256, 0, stream>>>(enc,  encB,  (long)BB * SS * K2H);
    f2bf<<<512,  256, 0, stream>>>(Wdec, WdecB, (long)HH * KU);
    f2bf<<<2048, 256, 0, stream>>>(Wvoc, WvocB, (long)VV * HH);
    prep_wg<<<2048, 256, 0, stream>>>(Wih, Whh, Wg);
    prep_bg<<<8,    256, 0, stream>>>(bih, bhh, bg);
    init_state<<<64, 256, 0, stream>>>(h0, c0, h, c, obf);

    // keys = enc @ W_att^T : M=4096 N=512 K=1024 -> 256x32 tiles
    wmma_gemm_bf16<<<1024, 256, 0, stream>>>(encB, K2H, WattB, K2H,
                                             keys, nullptr, nullptr, HH,
                                             nullptr, 256, 32, 0);

    // ---- sequential decode ----
    for (int t = 0; t < TT; ++t) {
        pack_xbar<<<BB, 256, 0, stream>>>(ids, emb, obf, h, xbar, t);
        // gates = x_bar @ Wg^T + bg : M=32 N=2048 K=1280 -> 2x128 tiles
        wmma_gemm_bf16<<<32, 256, 0, stream>>>(xbar, KG, Wg, KG,
                                               G, nullptr, nullptr, NG,
                                               bg, 2, 128, 0);
        lstm_cell<<<64, 256, 0, stream>>>(G, h, c);
        attn_ctx<<<BB, 128, 0, stream>>>(h, keys, enc, mask, u);
        // o = tanh(u @ Wdec^T + bdec) : M=32 N=512 K=1536 -> 2x32 tiles
        wmma_gemm_bf16<<<8, 256, 0, stream>>>(u, KU, WdecB, KU,
                                              nullptr, obf,
                                              oall + (long)t * BB * HH, HH,
                                              bdec, 2, 32, 1);
    }

    // logits = o_all @ W_vocab^T + b_vocab : M=4096 N=32000 K=512
    // 128 x 250 macro-tiles of 32x128
    wmma_gemm_vocab_lds<<<128 * (VV / NC), 256, 0, stream>>>(oall, WvocB, bvoc, out);
    // in-place log_softmax over vocab
    log_softmax_rows<<<BB * TT, 256, 0, stream>>>(out);
}
